// LocalMeanInpainter_9869834846597
// MI455X (gfx1250) — compile-verified, matrix operand order checked by reading, not run on hardware
//
#include <hip/hip_runtime.h>

// CDNA5 / gfx1250: wave32, WMMA 16x16x32 bf16 with f32 accumulate,
// async global->LDS staging (ASYNCcnt).
typedef __attribute__((ext_vector_type(16))) __bf16 v16bf;
typedef __attribute__((ext_vector_type(8)))  float  v8f;

#define IMG_H 512
#define IMG_W 512

// in-bounds count of a 15-wide window centered at t in [0,n)
__device__ __forceinline__ int span15(int t, int n) {
  int lo = t - 7; if (lo < 0) lo = 0;
  int hi = t + 7; if (hi > n - 1) hi = n - 1;
  return hi - lo + 1;
}

// low 32 bits of a flat pointer into LDS == LDS byte offset
__device__ __forceinline__ unsigned lds_off(const void* p) {
  return (unsigned)(size_t)p;
}

__global__ __launch_bounds__(256)
void inpaint_box15_wmma(const float* __restrict__ x,
                        const float* __restrict__ mask,
                        float* __restrict__ out) {
  // Staged input: LDS row r <-> image row R0-7+r, LDS col c <-> image col C0-8+c.
  // (one extra column of left halo so 4-float transfers are 16B aligned in both
  // global and LDS). Stride 84 floats keeps every row start 16B aligned and
  // spreads the 16 A-operand rows across banks (84 = 4*21, 21 coprime with 16).
  __shared__ float lds[80][84];

  const int tid  = threadIdx.x;
  const int lane = tid & 31;
  const int wv   = tid >> 5;        // 8 waves
  const int m    = lane & 15;       // row (A) / col (B,C,D) index within 16x16
  const int half = lane >> 4;       // lane group 0/1 (K striping)

  const int blk   = blockIdx.x;     // 96 planes * 64 tiles
  const int plane = blk >> 6;
  const int t64   = blk & 63;
  const int R0    = (t64 >> 3) * 64;
  const int C0    = (t64 & 7) * 64;

  const float* xp = x    + (size_t)plane * (IMG_H * IMG_W);
  const float* mp = mask + (size_t)plane * (IMG_H * IMG_W);
  float*       op = out  + (size_t)plane * (IMG_H * IMG_W);

  // A-operand loads touch LDS column 80 (always multiplied by a 0 band entry,
  // but NaN*0 = NaN in WMMA, so it must hold a finite value).
  if (tid < 80) lds[tid][80] = 0.0f;

  const bool interior = (R0 >= 64) && (R0 <= 384) && (C0 >= 64) && (C0 <= 384);
  if (interior) {
    // ---- async staging: 80 rows x 20 float4, global -> LDS, no VGPR landing ----
    const float* gbase = xp + (R0 - 7) * IMG_W + (C0 - 8);  // uniform -> SGPR pair
    for (int idx = tid; idx < 80 * 20; idx += 256) {
      int r = idx / 20, q = idx - r * 20;                   // q = float4 within row
      unsigned dsa  = lds_off(&lds[r][q * 4]);              // 16B aligned (84*4, 16*q)
      unsigned goff = (unsigned)((r * IMG_W + q * 4) * 4);  // 16B aligned byte offset
      asm volatile("global_load_async_to_lds_b128 %0, %1, %2"
                   :: "v"(dsa), "v"(goff), "s"(gbase) : "memory");
    }
    asm volatile("s_wait_asynccnt 0" ::: "memory");
  } else {
    // ---- border blocks: guarded scalar staging with zero padding ----
    for (int idx = tid; idx < 80 * 80; idx += 256) {
      int r = idx / 80, c = idx - r * 80;
      int gr = R0 - 7 + r, gc = C0 - 8 + c;
      float v = 0.0f;
      if ((unsigned)gr < IMG_H && (unsigned)gc < IMG_W) v = xp[gr * IMG_W + gc];
      lds[r][c] = v;
    }
  }
  __syncthreads();

  // ---- banded-ones matrix (exact in bf16) ----
  // Element j of the v16bf maps to K = half*8 + j (+8 for the upper 8 elems).
  // Same per-lane pattern serves as the A operand (16x32, row m) of the vertical
  // pass and the B operand (32x16, col n=m) of the horizontal pass; the band
  // condition is (m <= K <= m+14) in both roles.
  v16bf band;
#pragma unroll
  for (int j = 0; j < 16; ++j) {
    int k = half * 8 + j + ((j < 8) ? 0 : 8);
    band[j] = (k >= m && k <= m + 14) ? (__bf16)1.0f : (__bf16)0.0f;
  }

  // ---- 2 output tiles (16x16) per wave over the 64x64 region ----
  for (int tt = 0; tt < 2; ++tt) {
    const int t  = wv + tt * 8;         // 0..15
    const int lr = (t >> 2) * 16;       // local output row of tile
    const int lc = (t & 3) * 16;        // local output col of tile

    // ---------- horizontal pass: Mid(32x16) = In x Band ----------
    // A rows: LDS rows lr..lr+31 (image rows R0+lr-7..), K=0 at image col
    // C0+lc-7 which is LDS col lc+1.
    v8f mid_top = {};
    v8f mid_bot = {};
    {
      const float* rp = &lds[lr + m][lc + 1];
      v16bf ahi, alo;
#pragma unroll
      for (int j = 0; j < 16; ++j) {
        int k = half * 8 + j + ((j < 8) ? 0 : 8);
        float f = rp[k];
        __bf16 h = (__bf16)f;
        ahi[j] = h;
        alo[j] = (__bf16)(f - (float)h);
      }
      mid_top = __builtin_amdgcn_wmma_f32_16x16x32_bf16(
          false, ahi, false, band, (short)0, mid_top, false, false);
      mid_top = __builtin_amdgcn_wmma_f32_16x16x32_bf16(
          false, alo, false, band, (short)0, mid_top, false, false);
    }
    {
      const float* rp = &lds[lr + 16 + m][lc + 1];
      v16bf ahi, alo;
#pragma unroll
      for (int j = 0; j < 16; ++j) {
        int k = half * 8 + j + ((j < 8) ? 0 : 8);
        float f = rp[k];
        __bf16 h = (__bf16)f;
        ahi[j] = h;
        alo[j] = (__bf16)(f - (float)h);
      }
      mid_bot = __builtin_amdgcn_wmma_f32_16x16x32_bf16(
          false, ahi, false, band, (short)0, mid_bot, false, false);
      mid_bot = __builtin_amdgcn_wmma_f32_16x16x32_bf16(
          false, alo, false, band, (short)0, mid_bot, false, false);
    }

    // ---------- repack Mid (C/D f32 layout) as vertical B operand ----------
    // C/D layout: lane n holds col n, VGPR i = row i (lanes 0-15) / i+8
    // (lanes 16-31) -> identical to the 32x16 bf16 B-operand K striping.
    v16bf bhi, blo;
#pragma unroll
    for (int j = 0; j < 8; ++j) {
      float f = mid_top[j];
      __bf16 h = (__bf16)f;
      bhi[j] = h;
      blo[j] = (__bf16)(f - (float)h);
      float g = mid_bot[j];
      __bf16 h2 = (__bf16)g;
      bhi[8 + j] = h2;
      blo[8 + j] = (__bf16)(g - (float)h2);
    }

    // ---------- vertical pass: Out(16x16) = Band x Mid ----------
    v8f acc = {};
    acc = __builtin_amdgcn_wmma_f32_16x16x32_bf16(
        false, band, false, bhi, (short)0, acc, false, false);
    acc = __builtin_amdgcn_wmma_f32_16x16x32_bf16(
        false, band, false, blo, (short)0, acc, false, false);

    // ---------- epilogue: mean = boxsum/cnt ; blend with mask ----------
#pragma unroll
    for (int i = 0; i < 8; ++i) {
      int gr = R0 + lr + half * 8 + i;   // output row for this C/D VGPR
      int gc = C0 + lc + m;              // output col
      float cnt  = (float)(span15(gr, IMG_H) * span15(gc, IMG_W));
      float mean = acc[i] * __builtin_amdgcn_rcpf(cnt);
      float xv   = lds[lr + 7 + half * 8 + i][lc + 8 + m];
      float mk   = mp[gr * IMG_W + gc];
      op[gr * IMG_W + gc] = xv * mk + mean * (1.0f - mk);
    }
  }
}

extern "C" void kernel_launch(void* const* d_in, const int* in_sizes, int n_in,
                              void* d_out, int out_size, void* d_ws, size_t ws_size,
                              hipStream_t stream) {
  (void)in_sizes; (void)n_in; (void)out_size; (void)d_ws; (void)ws_size;
  const float* x    = (const float*)d_in[0];
  const float* mask = (const float*)d_in[1];
  float*       out  = (float*)d_out;

  // 32*3 = 96 planes, each 512x512 split into 8x8 tiles of 64x64.
  dim3 grid(96 * 64);
  dim3 block(256);
  inpaint_box15_wmma<<<grid, block, 0, stream>>>(x, mask, out);
}
